// ParallaxAttention_60335700574394
// MI455X (gfx1250) — matrix operand order repaired and checked by reading, not compiled, not run on hardware
//
#include <hip/hip_runtime.h>

// ----------------------------------------------------------------------------
// ParallaxAttention for MI455X (gfx1250, wave32, WMMA).
// - All matrix math: v_wmma_f32_16x16x32_bf16 (fp32 accumulate)
// - A-tile staging via global_load_async_to_lds_b128 (ASYNCcnt path)
// - K/V prefetch via global_prefetch_b8
// Workspace needed: ~126 MB (lifetime-aliased regions, see kernel_launch).
// ----------------------------------------------------------------------------

#define SEQ   2048
#define HDIM  4096
#define NQH   32
#define NKVH  8
#define HDSZ  128
#define DQ    (NQH * HDSZ)    // 4096
#define DKV   (NKVH * HDSZ)   // 1024

typedef __attribute__((ext_vector_type(16))) __bf16        v16bf;
typedef __attribute__((ext_vector_type(8)))  float         v8f;
typedef __attribute__((ext_vector_type(4)))  unsigned int  v4u;

union bfrag {
  v16bf          bf;      // per-lane 16 bf16 = 8 VGPRs (WMMA A/B operand)
  v4u            q[2];    // two 16-byte chunks
  unsigned short u[16];
};

__device__ __forceinline__ unsigned short f32_to_bf16(float f) {
  unsigned int x = __float_as_uint(f);
  x += 0x7FFFu + ((x >> 16) & 1u);          // round-to-nearest-even
  return (unsigned short)(x >> 16);
}
__device__ __forceinline__ float bf16_to_f32(unsigned short u) {
  return __uint_as_float((unsigned int)u << 16);
}

// Async global->LDS copy (GLOBAL_LOAD_ASYNC_TO_LDS_B128, GVS mode).
// lds_off: LDS byte offset (low 32 bits of a flat shared pointer),
// goff: 32-bit per-lane byte offset, sbase: SGPR 64-bit base.
__device__ __forceinline__ void async_copy_b128(unsigned lds_off, unsigned goff,
                                                const void* sbase) {
  asm volatile("global_load_async_to_lds_b128 %0, %1, %2"
               :
               : "v"(lds_off), "v"(goff), "s"(sbase)
               : "memory");
}
__device__ __forceinline__ void wait_asynccnt0() {
  asm volatile("s_wait_asynccnt 0x0" ::: "memory");
}

// ---------------------------- fp32 -> bf16 ----------------------------------
__global__ void cvt_f32_bf16(const float* __restrict__ src,
                             unsigned short* __restrict__ dst, int n) {
  int i = blockIdx.x * blockDim.x + threadIdx.x;
  if (i < n) dst[i] = f32_to_bf16(src[i]);
}

// ---------------------------- RoPE (bf16 in/out) ----------------------------
// layout [s][h][d], d = 128; rotate_half convention of the reference.
__global__ void rope_bf16(const unsigned short* __restrict__ src,
                          unsigned short* __restrict__ dst,
                          const float* __restrict__ cosb,
                          const float* __restrict__ sinb,
                          int nh, int n) {
  int i = blockIdx.x * blockDim.x + threadIdx.x;
  if (i >= n) return;
  int d  = i & (HDSZ - 1);
  int sh = i >> 7;            // s*nh + h
  int s  = sh / nh;
  float x   = bf16_to_f32(src[i]);
  float rot = (d < HDSZ / 2) ? -bf16_to_f32(src[i + HDSZ / 2])
                             :  bf16_to_f32(src[i - HDSZ / 2]);
  float v = x * cosb[s * HDSZ + d] + rot * sinb[s * HDSZ + d];
  dst[i] = f32_to_bf16(v);
}

// ---------------------------- bf16 WMMA GEMM --------------------------------
// C(MxN) = A(bf16, MxK row-major) * B(bf16, KxN row-major); C f32 or bf16.
// Workgroup: 256 thr = 8 waves; tile 128x128, K-step 32; wave = 32x64 subtile
// (8 WMMAs / wave / K-step). A staged by async global->LDS copies; B staged
// transposed [n][k] through VGPRs so fragment reads are contiguous 16B ds
// loads. 40-ushort (80B) padded stride: 16B-aligned, 16 distinct banks.
#define BM 128
#define BN 128
#define BK 32
#define LPAD 40

template <int N, int K, bool BF16OUT>
__global__ __launch_bounds__(256)
void gemm_bf16_wmma(const unsigned short* __restrict__ A,
                    const unsigned short* __restrict__ B,
                    void* __restrict__ Cout) {
  __shared__ alignas(16) unsigned short As[BM * LPAD];
  __shared__ alignas(16) unsigned short Bs[BN * LPAD];

  const int tid  = threadIdx.x;
  const int lane = tid & 31;
  const int wid  = tid >> 5;
  const int wm   = wid >> 1;            // 0..3 : 32-row subtile
  const int wn   = wid & 1;             // 0..1 : 64-col half
  const int fr   = lane & 15;
  const int hi   = lane >> 4;
  const int kh   = hi * 16;             // A-frag K split

  const int m0 = blockIdx.y * BM;
  const int n0 = blockIdx.x * BN;

  // A async staging: 128x32 tile = 512 16B chunks, 2 per thread
  const int ar = tid >> 2;              // rows ar and ar+64
  const int ac = (tid & 3) * 8;
  const unsigned lds_a0 = (unsigned)(size_t)(As + (ar * LPAD + ac));
  const unsigned lds_a1 = (unsigned)(size_t)(As + ((ar + 64) * LPAD + ac));
  const unsigned ga0 = (unsigned)((((size_t)(m0 + ar)) * K + ac) * 2);
  const unsigned ga1 = (unsigned)((((size_t)(m0 + ar + 64)) * K + ac) * 2);

  const int brow = tid >> 4;            // 2 k-rows x 8 ushorts per thread
  const int bcol = (tid & 15) * 8;

  v8f acc[2][4] = {};

  for (int k0 = 0; k0 < K; k0 += BK) {
    // A: async global->LDS (no VGPR round-trip), overlaps with B staging
    async_copy_b128(lds_a0, ga0 + (unsigned)(k0 * 2), A);
    async_copy_b128(lds_a1, ga1 + (unsigned)(k0 * 2), A);

    // B: stage transposed [n][k]
#pragma unroll
    for (int rr = 0; rr < 2; ++rr) {
      int kk = brow + rr * 16;
      union { v4u v; unsigned short u[8]; } t;
      t.v = *(const v4u*)(B + (size_t)(k0 + kk) * N + n0 + bcol);
#pragma unroll
      for (int e = 0; e < 8; ++e) Bs[(bcol + e) * LPAD + kk] = t.u[e];
    }
    wait_asynccnt0();            // own async copies landed in LDS
    __syncthreads();             // + everyone's (dscnt waited by barrier)

    bfrag a[2];
#pragma unroll
    for (int ss = 0; ss < 2; ++ss) {
      const unsigned short* ap = As + (32 * wm + 16 * ss + fr) * LPAD;
      a[ss].q[0] = *(const v4u*)(ap + kh);
      a[ss].q[1] = *(const v4u*)(ap + kh + 8);
    }
#pragma unroll
    for (int ct = 0; ct < 4; ++ct) {
      bfrag b;   // column 64*wn+16*ct+fr, K = hi*16..hi*16+15 contiguous
      const unsigned short* bp = Bs + (64 * wn + 16 * ct + fr) * LPAD + hi * 16;
      b.q[0] = *(const v4u*)(bp);
      b.q[1] = *(const v4u*)(bp + 8);
#pragma unroll
      for (int ss = 0; ss < 2; ++ss)
        acc[ss][ct] = __builtin_amdgcn_wmma_f32_16x16x32_bf16(
            false, a[ss].bf, false, b.bf, (short)0, acc[ss][ct], false, false);
    }
    __syncthreads();
  }

  // C layout: VGPR i -> row i (lanes 0-15) / i+8 (lanes 16-31), col = lane&15
#pragma unroll
  for (int ss = 0; ss < 2; ++ss) {
    const int rbase = m0 + 32 * wm + 16 * ss + hi * 8;
#pragma unroll
    for (int ct = 0; ct < 4; ++ct) {
      const int col = n0 + 64 * wn + 16 * ct + fr;
#pragma unroll
      for (int i = 0; i < 8; ++i) {
        if constexpr (BF16OUT)
          ((unsigned short*)Cout)[(size_t)(rbase + i) * N + col] =
              f32_to_bf16(acc[ss][ct][i]);
        else
          ((float*)Cout)[(size_t)(rbase + i) * N + col] = acc[ss][ct][i];
      }
    }
  }
}

// ---------------------------- flash attention -------------------------------
// grid = (32 heads, 32 q-tiles of 64 rows), block = 128 thr (4 waves).
// Each wave: 16 query rows x d=128, online softmax, 32-key tiles.
__global__ __launch_bounds__(128)
void flash_attn_wmma(const unsigned short* __restrict__ Qb,
                     const unsigned short* __restrict__ Kb,
                     const unsigned short* __restrict__ Vb,
                     unsigned short* __restrict__ Ob) {
  __shared__ alignas(16) unsigned short Pl[4][16 * 32];   // per-wave P staging

  const int h     = blockIdx.x;
  const int hkv   = h >> 2;                 // GQA group of 4
  const int wid   = threadIdx.x >> 5;
  const int lane  = threadIdx.x & 31;
  const int fr    = lane & 15;
  const int hi    = lane >> 4;
  const int kh    = hi * 16;
  const int qbase = blockIdx.y * 64 + wid * 16;

  // Q as 4 A-fragments over d (4 x 16x32)
  bfrag qf[4];
  const unsigned short* qrow = Qb + (size_t)(qbase + fr) * DQ + h * HDSZ;
#pragma unroll
  for (int c = 0; c < 4; ++c) {
    qf[c].q[0] = *(const v4u*)(qrow + 32 * c + kh);
    qf[c].q[1] = *(const v4u*)(qrow + 32 * c + kh + 8);
  }

  v8f o[8] = {};                             // O accum 16x128
  float mrow[8], lrow[8];
#pragma unroll
  for (int i = 0; i < 8; ++i) { mrow[i] = -3.0e38f; lrow[i] = 0.f; }

  const float scale = 0.08838834764831845f;  // 1/sqrt(128)
  unsigned short* pl = &Pl[wid][0];

  for (int kb = 0; kb < qbase + 16; kb += 32) {     // causal key range
    // prefetch next tile's K/V rows into cache while this tile computes
    if (kb + 32 < qbase + 16) {
      __builtin_prefetch(Kb + (size_t)(kb + 32 + fr) * DKV + hkv * HDSZ, 0, 3);
      __builtin_prefetch(Vb + (size_t)(kb + 32 + fr) * DKV + hkv * HDSZ, 0, 3);
    }

    // S = Q * K^T for two 16-key column tiles
    v8f sa = {}, sb = {};
#pragma unroll
    for (int c = 0; c < 4; ++c) {
      bfrag ka, kbf;
      const unsigned short* kra =
          Kb + (size_t)(kb + fr) * DKV + hkv * HDSZ + 32 * c + hi * 16;
      ka.q[0]  = *(const v4u*)(kra);
      ka.q[1]  = *(const v4u*)(kra + 8);
      const unsigned short* krb =
          Kb + (size_t)(kb + 16 + fr) * DKV + hkv * HDSZ + 32 * c + hi * 16;
      kbf.q[0] = *(const v4u*)(krb);
      kbf.q[1] = *(const v4u*)(krb + 8);
      sa = __builtin_amdgcn_wmma_f32_16x16x32_bf16(false, qf[c].bf, false, ka.bf,
                                                   (short)0, sa, false, false);
      sb = __builtin_amdgcn_wmma_f32_16x16x32_bf16(false, qf[c].bf, false, kbf.bf,
                                                   (short)0, sb, false, false);
    }

    // online softmax; row r lives in acc element i (=r or r-8) across 16 lanes
#pragma unroll
    for (int i = 0; i < 8; ++i) {
      const int row = qbase + i + 8 * hi;
      float va = sa[i] * scale;
      float vb = sb[i] * scale;
      if (kb + fr > row)      va = -3.0e38f;          // causal mask
      if (kb + 16 + fr > row) vb = -3.0e38f;

      float tm = fmaxf(va, vb);
      tm = fmaxf(tm, __shfl_xor(tm, 1, 32));          // stays inside 16-lane half
      tm = fmaxf(tm, __shfl_xor(tm, 2, 32));
      tm = fmaxf(tm, __shfl_xor(tm, 4, 32));
      tm = fmaxf(tm, __shfl_xor(tm, 8, 32));
      float mnew  = fmaxf(mrow[i], tm);
      float alpha = __expf(mrow[i] - mnew);
      float pa = __expf(va - mnew);
      float pb = __expf(vb - mnew);
      mrow[i] = mnew;
      float ps = pa + pb;
      ps += __shfl_xor(ps, 1, 32);
      ps += __shfl_xor(ps, 2, 32);
      ps += __shfl_xor(ps, 4, 32);
      ps += __shfl_xor(ps, 8, 32);
      lrow[i] = lrow[i] * alpha + ps;
#pragma unroll
      for (int c = 0; c < 8; ++c) o[c][i] *= alpha;   // rescale O row
      const int prow = i + 8 * hi;
      pl[prow * 32 + fr]      = f32_to_bf16(pa);      // P -> LDS (C-layout)
      pl[prow * 32 + 16 + fr] = f32_to_bf16(pb);
    }

    // P back from LDS in A-fragment layout (layout transpose; DS ops are
    // in-order within a wave)
    bfrag pf;
    pf.q[0] = *(const v4u*)(pl + fr * 32 + kh);
    pf.q[1] = *(const v4u*)(pl + fr * 32 + kh + 8);

    // O += P * V  (8 d-tiles of 16)
#pragma unroll
    for (int c = 0; c < 8; ++c) {
      bfrag vf;                                       // B frag: col d=16c+fr,
#pragma unroll                                        // K = hi*16 + 0..15
      for (int e = 0; e < 16; ++e)
        vf.u[e] = Vb[(size_t)(kb + hi * 16 + e) * DKV + hkv * HDSZ + 16 * c + fr];
      o[c] = __builtin_amdgcn_wmma_f32_16x16x32_bf16(false, pf.bf, false, vf.bf,
                                                     (short)0, o[c], false, false);
    }
  }

  // normalize and store bf16 attention output [s][h][d]
#pragma unroll
  for (int i = 0; i < 8; ++i) {
    float inv = 1.0f / lrow[i];
    const size_t row = qbase + i + 8 * hi;
#pragma unroll
    for (int c = 0; c < 8; ++c)
      Ob[row * DQ + h * HDSZ + 16 * c + fr] = f32_to_bf16(o[c][i] * inv);
  }
}

// ---------------------------- launcher --------------------------------------
extern "C" void kernel_launch(void* const* d_in, const int* in_sizes, int n_in,
                              void* d_out, int out_size, void* d_ws, size_t ws_size,
                              hipStream_t stream) {
  (void)in_sizes; (void)n_in; (void)out_size; (void)ws_size;

  const float* hid  = (const float*)d_in[0];
  const float* cosb = (const float*)d_in[1];
  const float* sinb = (const float*)d_in[2];
  const float* wq   = (const float*)d_in[3];
  const float* wk   = (const float*)d_in[4];
  const float* wv   = (const float*)d_in[5];
  const float* wo   = (const float*)d_in[6];
  float* out = (float*)d_out;
  char*  ws  = (char*)d_ws;

  // workspace layout (bytes); total ~126 MB. Aliased regions have disjoint
  // lifetimes in stream order.
  const size_t OFF_HID = 0;
  const size_t OFF_WQ  = OFF_HID + (size_t)SEQ * HDIM * 2;   // hidden bf16
  const size_t OFF_WK  = OFF_WQ  + (size_t)HDIM * DQ  * 2;
  const size_t OFF_WV  = OFF_WK  + (size_t)HDIM * DKV * 2;
  const size_t OFF_WO  = OFF_WV  + (size_t)HDIM * DKV * 2;
  const size_t OFF_QP  = OFF_WO  + (size_t)DQ * HDIM * 2;    // Q pre-rope bf16
  const size_t OFF_KP  = OFF_QP  + (size_t)SEQ * DQ  * 2;    // K pre-rope bf16
  const size_t OFF_VB  = OFF_KP  + (size_t)SEQ * DKV * 2;    // V bf16 (final)
  const size_t OFF_QB  = OFF_WQ;   // rope'd Q over dead wq
  const size_t OFF_KB  = OFF_WK;   // rope'd K over dead wk
  const size_t OFF_ATT = OFF_QP;   // attn out over dead pre-rope Q

  unsigned short* hb   = (unsigned short*)(ws + OFF_HID);
  unsigned short* wqb  = (unsigned short*)(ws + OFF_WQ);
  unsigned short* wkb  = (unsigned short*)(ws + OFF_WK);
  unsigned short* wvb  = (unsigned short*)(ws + OFF_WV);
  unsigned short* wob  = (unsigned short*)(ws + OFF_WO);
  unsigned short* qp   = (unsigned short*)(ws + OFF_QP);
  unsigned short* kp   = (unsigned short*)(ws + OFF_KP);
  unsigned short* vbb  = (unsigned short*)(ws + OFF_VB);
  unsigned short* qbb  = (unsigned short*)(ws + OFF_QB);
  unsigned short* kbb  = (unsigned short*)(ws + OFF_KB);
  unsigned short* attb = (unsigned short*)(ws + OFF_ATT);

  auto cvt = [&](const float* s, unsigned short* d, int n) {
    cvt_f32_bf16<<<(n + 255) / 256, 256, 0, stream>>>(s, d, n);
  };

  // 1) fp32 -> bf16 once per call
  cvt(hid, hb, SEQ * HDIM);
  cvt(wq, wqb, HDIM * DQ);
  cvt(wk, wkb, HDIM * DKV);
  cvt(wv, wvb, HDIM * DKV);
  cvt(wo, wob, DQ * HDIM);

  // 2) Q/K/V projections (bf16 WMMA, fp32 accum, bf16 out)
  gemm_bf16_wmma<DQ,  HDIM, true><<<dim3(DQ / BN,  SEQ / BM), 256, 0, stream>>>(hb, wqb, qp);
  gemm_bf16_wmma<DKV, HDIM, true><<<dim3(DKV / BN, SEQ / BM), 256, 0, stream>>>(hb, wkb, kp);
  gemm_bf16_wmma<DKV, HDIM, true><<<dim3(DKV / BN, SEQ / BM), 256, 0, stream>>>(hb, wvb, vbb);

  // 3) RoPE (bf16 in/out)
  rope_bf16<<<(SEQ * DQ  + 255) / 256, 256, 0, stream>>>(qp, qbb, cosb, sinb, NQH,  SEQ * DQ);
  rope_bf16<<<(SEQ * DKV + 255) / 256, 256, 0, stream>>>(kp, kbb, cosb, sinb, NKVH, SEQ * DKV);

  // 4) causal GQA flash attention
  flash_attn_wmma<<<dim3(NQH, SEQ / 64), 128, 0, stream>>>(qbb, kbb, vbb, attb);

  // 5) output projection -> fp32 d_out
  gemm_bf16_wmma<HDIM, DQ, false><<<dim3(HDIM / BN, SEQ / BM), 256, 0, stream>>>(attb, wob, out);
}
